// metapath2vec_model_86835648790550
// MI455X (gfx1250) — matrix operand order repaired and checked by reading, not compiled, not run on hardware
//
#include <hip/hip_runtime.h>

typedef __attribute__((ext_vector_type(16))) _Float16 v16h;
typedef __attribute__((ext_vector_type(8)))  float    v8f;
typedef __attribute__((ext_vector_type(4)))  unsigned int u32x4;
typedef __attribute__((ext_vector_type(8)))  int          i32x8;
typedef __attribute__((ext_vector_type(4)))  int          i32x4;

#define EMBED_D 128
#define ROW_F   130          // padded LDS row stride (floats): 512B data + 8B pad -> conflict-free
#define N_NEG   5
#define N_TGT   (N_NEG + 1)
#define WAVESZ  32
#define WAVES_PER_BLOCK 2
#define THREADS (WAVES_PER_BLOCK * WAVESZ)
#define PAIRS_PER_WAVE 16
#define PAIRS_PER_BLOCK (WAVES_PER_BLOCK * PAIRS_PER_WAVE)
#define TILE_F (PAIRS_PER_WAVE * ROW_F)      // floats per 16-row LDS tile
#define WAVE_F (3 * TILE_F)                  // A + B0 + B1 per wave

union AFrag { v16h h; unsigned int u[8]; };

__global__ void sgns_zero(float* ws) { ws[0] = 0.0f; }

// TDM gather-mode: pull 8 indexed embedding rows (f32, 128 wide) into LDS with
// a 2-dword pad after every 128 dwords (520B LDS row stride).
__device__ inline void tdm_gather8(const float* __restrict__ emb, unsigned lds_off,
                                   int i0, int i1, int i2, int i3,
                                   int i4, int i5, int i6, int i7)
{
  unsigned long long ga = (unsigned long long)(uintptr_t)emb;
  u32x4 g0; i32x8 g1; i32x4 g2, g3;
  g0[0] = 0xC0000001u;                                  // count=1 | 32b idx | gather_mode
  g0[1] = lds_off;                                      // LDS byte address
  g0[2] = (unsigned)ga;                                 // global_addr[31:0]
  g0[3] = ((unsigned)(ga >> 32) & 0x01FFFFFFu) | 0x80000000u; // addr[56:32] | type=2
  g1[0] = 0x03920000;   // data_size=4B | pad_enable | pad_interval=128dw | pad_amount=2dw
  g1[1] = (int)(EMBED_D << 16);                         // tensor_dim0 = 128 (lo16)
  g1[2] = (int)(0x4240u << 16);                         // tensor_dim1 lo16 (1e6 rows)
  g1[3] = (int)((EMBED_D << 16) | 0xF);                 // tile_dim0=128 | tensor_dim1 hi16
  g1[4] = 8;                                            // tile_dim1 = #gather indices
  g1[5] = EMBED_D;                                      // tensor_dim0_stride = 128
  g1[6] = 0;
  g1[7] = 0;
  g2[0] = i0; g2[1] = i1; g2[2] = i2; g2[3] = i3;       // row indices 0..3
  g3[0] = i4; g3[1] = i5; g3[2] = i6; g3[3] = i7;       // row indices 4..7
  i32x8 gz = {0, 0, 0, 0, 0, 0, 0, 0};                  // 6-arg form: zero-filled group
  __builtin_amdgcn_tensor_load_to_lds(g0, g1, g2, g3, gz, 0);
}

// Stage 16 rows (two TDM gather ops) whose indices live in lanes 0..15 of `lanereg`.
__device__ inline void tdm_stage16(const float* __restrict__ emb, unsigned lds_off,
                                   int lanereg)
{
  int ix[16];
  #pragma unroll
  for (int r = 0; r < 16; ++r)
    ix[r] = __builtin_amdgcn_readfirstlane(__shfl(lanereg, r, WAVESZ));
  tdm_gather8(emb, lds_off,
              ix[0], ix[1], ix[2], ix[3], ix[4], ix[5], ix[6], ix[7]);
  tdm_gather8(emb, lds_off + 8u * ROW_F * 4u,
              ix[8], ix[9], ix[10], ix[11], ix[12], ix[13], ix[14], ix[15]);
}

__global__ __launch_bounds__(THREADS) void sgns_wmma(
    const float* __restrict__ emb,
    const int*  __restrict__ centers,
    const int*  __restrict__ contexts,
    const int*  __restrict__ negs,
    float* __restrict__ ws, int P)
{
  extern __shared__ float smem[];

  const int lane = threadIdx.x & (WAVESZ - 1);
  const int wave = threadIdx.x >> 5;
  const int p0   = ((int)blockIdx.x * WAVES_PER_BLOCK + wave) * PAIRS_PER_WAVE;

  float* Abuf = smem + wave * WAVE_F;
  float* B0   = Abuf + TILE_F;
  float* B1   = B0 + TILE_F;
  const unsigned a_off  = (unsigned)__builtin_amdgcn_readfirstlane(
                              (int)(unsigned)(uintptr_t)(void*)Abuf);
  const unsigned b_off0 = a_off + (unsigned)(TILE_F * 4);
  const unsigned b_off1 = b_off0 + (unsigned)(TILE_F * 4);

  const int myp = min(p0 + (lane & 15), P - 1);

  // --- async: gather A (centers) then B(t=0) (contexts); TENSORcnt tracks both ---
  int cidx = (lane < PAIRS_PER_WAVE) ? centers[myp]  : 0;
  tdm_stage16(emb, a_off, cidx);
  int tidx = (lane < PAIRS_PER_WAVE) ? contexts[myp] : 0;
  tdm_stage16(emb, b_off0, tidx);

  __builtin_amdgcn_s_wait_tensorcnt(2);   // A's two ops complete (in-order)
  asm volatile("" ::: "memory");

  // ---- A fragments: 4 chunks of 16x32 f16 (ISA 16-bit A layout), cvt from f32 LDS ----
  AFrag a[4];
  {
    const int m  = lane & 15;
    const int kh = (lane >> 4) * 8;
    #pragma unroll
    for (int c = 0; c < 4; ++c) {
      #pragma unroll
      for (int i = 0; i < 8; ++i) {
        int k = 32 * c + ((i >= 4) ? 16 : 0) + kh + (i & 3) * 2;
        float2 f = *reinterpret_cast<const float2*>(Abuf + m * ROW_F + k);
        union { _Float16 h[2]; unsigned u; } q;
        q.h[0] = (_Float16)f.x; q.h[1] = (_Float16)f.y;
        a[c].u[i] = q.u;
      }
    }
  }

  // diagonal ownership per ISA C/D layout
  int vsel = -1, pown = -1;
  if (lane < 8)        { vsel = lane;      pown = lane;      }
  else if (lane >= 24) { vsel = lane - 24; pown = lane - 16; }

  float lsum = 0.0f;

  #pragma unroll
  for (int t = 0; t < N_TGT; ++t) {
    asm volatile("" ::: "memory");
    if (t < N_TGT - 1) {
      // prefetch next target (neg t) into the other buffer, then wait for B(t)
      int nidx = (lane < PAIRS_PER_WAVE) ? negs[myp * N_NEG + t] : 0;
      tdm_stage16(emb, (t & 1) ? b_off0 : b_off1, nidx);
      __builtin_amdgcn_s_wait_tensorcnt(2);
    } else {
      __builtin_amdgcn_s_wait_tensorcnt(0);
    }
    asm volatile("" ::: "memory");

    const float* Bb = (t & 1) ? B1 : B0;
    v8f acc = {};
    {
      const int n  = lane & 15;
      const int kb = (lane >= 16) ? 8 : 0;
      #pragma unroll
      for (int c = 0; c < 4; ++c) {
        AFrag b;
        #pragma unroll
        for (int i = 0; i < 8; ++i) {
          int k = 32 * c + ((i >= 4) ? 16 : 0) + kb + (i & 3) * 2;
          float2 f = *reinterpret_cast<const float2*>(Bb + n * ROW_F + k);
          union { _Float16 h[2]; unsigned u; } q;
          q.h[0] = (_Float16)f.x; q.h[1] = (_Float16)f.y;
          b.u[i] = q.u;
        }
        acc = __builtin_amdgcn_wmma_f32_16x16x32_f16(
            false, a[c].h, false, b.h, (short)0, acc, false, false);
      }
    }

    // diagonal -> per-pair score -> SGNS softplus terms
    if (vsel >= 0) {
      float s = acc[0];
      #pragma unroll
      for (int v = 1; v < 8; ++v) s = (vsel == v) ? acc[v] : s;
      if (p0 + pown < P) {
        float z  = (t == 0) ? -s : s;   // pos: softplus(-s); neg: softplus(+s)
        float sp = fmaxf(z, 0.0f) + __logf(1.0f + __expf(-fabsf(z)));
        lsum += sp;
      }
    }
  }

  // wave reduction; one atomic per wave
  #pragma unroll
  for (int off = 16; off >= 1; off >>= 1)
    lsum += __shfl_xor(lsum, off, WAVESZ);
  if (lane == 0) atomicAdd(ws, lsum);
}

__global__ void sgns_final(const float* __restrict__ ws, float* __restrict__ out,
                           float inv_p) {
  out[0] = ws[0] * inv_p;
}

extern "C" void kernel_launch(void* const* d_in, const int* in_sizes, int n_in,
                              void* d_out, int out_size, void* d_ws, size_t ws_size,
                              hipStream_t stream) {
  (void)n_in; (void)out_size; (void)ws_size;
  const float* emb      = (const float*)d_in[0];
  const int*   centers  = (const int*)d_in[1];
  const int*   contexts = (const int*)d_in[2];
  const int*   negs     = (const int*)d_in[3];
  float* ws  = (float*)d_ws;
  float* out = (float*)d_out;

  const int P = in_sizes[1];                                      // 248320
  const int blocks = (P + PAIRS_PER_BLOCK - 1) / PAIRS_PER_BLOCK; // 7760
  const size_t shmem = (size_t)WAVES_PER_BLOCK * WAVE_F * sizeof(float); // 49920 B

  sgns_zero <<<1, 1, 0, stream>>>(ws);
  sgns_wmma <<<blocks, THREADS, shmem, stream>>>(emb, centers, contexts, negs, ws, P);
  sgns_final<<<1, 1, 0, stream>>>(ws, out, 1.0f / (float)P);
}
